// Attention_77876347011151
// MI455X (gfx1250) — matrix operand order repaired and checked by reading, not compiled, not run on hardware
//
#include <hip/hip_runtime.h>
#include <hip/hip_bf16.h>
#include <stdint.h>

#define HEADS 12
#define DIM 768
#define DIM_HEAD 64
#define B_SZ 8
#define N_SEQ 1025
#define MTOT (B_SZ * N_SEQ)          /* 8200 */
#define NPAD 1056                    /* 33 * 32 key padding */
#define MTILES32 ((MTOT + 31) / 32)  /* 257: 32-row GEMM tiles */
#define LN_EPS 1e-5f
#define ATT_SCALE 0.125f             /* 1/sqrt(64) */

typedef __attribute__((ext_vector_type(16))) __bf16 v16bf;
typedef __attribute__((ext_vector_type(8)))  float  v8f;
typedef __attribute__((ext_vector_type(4)))  unsigned int v4u;

union FragU { v16bf v; v4u q[2]; };

__device__ __forceinline__ v8f zero8() { v8f z = {0.f, 0.f, 0.f, 0.f, 0.f, 0.f, 0.f, 0.f}; return z; }

__device__ __forceinline__ unsigned short f2bf(float f) {
  union { float f; unsigned int u; } x; x.f = f;
  unsigned int r = x.u + 0x7FFFu + ((x.u >> 16) & 1u);
  return (unsigned short)(r >> 16);
}

// D = A*B + C, bf16 inputs, f32 accumulate (CDNA5 16x16x32)
__device__ __forceinline__ v8f wmma_bf16(const FragU& a, const FragU& b, v8f c) {
  return __builtin_amdgcn_wmma_f32_16x16x32_bf16(false, a.v, false, b.v, (short)0, c,
                                                 false, false);
}

// A-fragment: lane half 0 holds K 0..15, half 1 holds 16..31.
// p must already point at rowBase + chunkBase + half*16 (16B aligned).
__device__ __forceinline__ void load_a2(FragU& f, const unsigned short* p) {
  f.q[0] = *(const v4u*)p;
  f.q[1] = *(const v4u*)(p + 8);
}
// B-fragment: lane holds 16 K-consecutive bf16 for one column (32B contiguous).
__device__ __forceinline__ void load_pk(FragU& f, const unsigned short* p) {
  f.q[0] = ((const v4u*)p)[0];
  f.q[1] = ((const v4u*)p)[1];
}

__device__ __forceinline__ int imin(int a, int b) { return a < b ? a : b; }

// ---------------------------------------------------------------------------
// Pack f32 weight matrix (DIM x ncols, row-major) into B-fragment-major bf16:
// dst[((kt*ntiles + nt)*32 + lane)*16 + e] = W[kt*32 + (lane/16)*16 + e][nt*16 + lane%16]
// ---------------------------------------------------------------------------
__global__ void __launch_bounds__(256)
pack_w_kernel(const float* __restrict__ W, unsigned short* __restrict__ dst, int ncols) {
  int t = blockIdx.x * 256 + threadIdx.x;
  if (t >= DIM * ncols) return;
  int e = t & 15;
  int lane = (t >> 4) & 31;
  int frag = t >> 9;
  int ntiles = ncols >> 4;
  int nt = frag % ntiles;
  int kt = frag / ntiles;
  int k = kt * 32 + (lane >> 4) * 16 + e;
  int n = nt * 16 + (lane & 15);
  dst[t] = f2bf(W[(size_t)k * ncols + n]);
}

// ---------------------------------------------------------------------------
// LayerNorm + f32 -> bf16 cast: one block (256 thr) per row of x
// ---------------------------------------------------------------------------
__global__ void __launch_bounds__(256)
ln_kernel(const float* __restrict__ x, const float* __restrict__ gamma,
          const float* __restrict__ beta, unsigned short* __restrict__ xn) {
  __shared__ float rs[8], rs2[8];
  __shared__ float mu_s, rstd_s;
  const int row = blockIdx.x;
  const int tid = threadIdx.x;
  const float* xr = x + (size_t)row * DIM;
  float v0 = xr[tid], v1 = xr[tid + 256], v2 = xr[tid + 512];
  float s = v0 + v1 + v2;
  float s2 = v0 * v0 + v1 * v1 + v2 * v2;
#pragma unroll
  for (int o = 16; o > 0; o >>= 1) {
    s += __shfl_xor(s, o, 32);
    s2 += __shfl_xor(s2, o, 32);
  }
  if ((tid & 31) == 0) { rs[tid >> 5] = s; rs2[tid >> 5] = s2; }
  __syncthreads();
  if (tid == 0) {
    float S = 0.f, S2 = 0.f;
#pragma unroll
    for (int i = 0; i < 8; ++i) { S += rs[i]; S2 += rs2[i]; }
    float mu = S / (float)DIM;
    float var = S2 / (float)DIM - mu * mu;
    mu_s = mu;
    rstd_s = rsqrtf(var + LN_EPS);
  }
  __syncthreads();
  const float mu = mu_s, rstd = rstd_s;
  unsigned short* orow = xn + (size_t)row * DIM;
  orow[tid]       = f2bf((v0 - mu) * rstd * gamma[tid]       + beta[tid]);
  orow[tid + 256] = f2bf((v1 - mu) * rstd * gamma[tid + 256] + beta[tid + 256]);
  orow[tid + 512] = f2bf((v2 - mu) * rstd * gamma[tid + 512] + beta[tid + 512]);
}

// ---------------------------------------------------------------------------
// QKV GEMM (WMMA bf16) with fused RoPE + head-split epilogue.
// Grid (257, 9), block 128 (4 waves). Wave handles a 32x64 tile == one head
// section of one of {q,k,v} for 32 token rows: 8 WMMAs per K-chunk against
// 4 shared B-fragments. A rows are index-clamped (no exec divergence); rows
// >= MTOT produce garbage that is never stored. Writes Qp/Kp (token-major
// bf16) and Vt (feature-major bf16), padded to NPAD tokens per (b,h).
// ---------------------------------------------------------------------------
__global__ void __launch_bounds__(128)
qkv_kernel(const unsigned short* __restrict__ xn, const unsigned short* __restrict__ wq,
           const float* __restrict__ coords, unsigned short* __restrict__ Qp,
           unsigned short* __restrict__ Kp, unsigned short* __restrict__ Vt) {
  __shared__ alignas(16) float stile[4][32 * DIM_HEAD];
  const int w = threadIdx.x >> 5, lane = threadIdx.x & 31;
  const int hv = lane >> 4, r = lane & 15;
  const int sec = blockIdx.y * 4 + w;        // 0..35
  const int cbase = sec * 64;
  const int mbase = blockIdx.x * 32;

  v8f acc[2][4];
#pragma unroll
  for (int g = 0; g < 2; ++g)
#pragma unroll
    for (int t = 0; t < 4; ++t) acc[g][t] = zero8();

  const unsigned short* arow0 = xn + (size_t)imin(mbase + r,      MTOT - 1) * DIM;
  const unsigned short* arow1 = xn + (size_t)imin(mbase + 16 + r, MTOT - 1) * DIM;

  for (int kt = 0; kt < DIM / 32; ++kt) {
    FragU af0, af1;
    load_a2(af0, arow0 + kt * 32 + hv * 16);
    load_a2(af1, arow1 + kt * 32 + hv * 16);
    const unsigned short* fb = wq + ((size_t)kt * 144 + (cbase >> 4)) * 512 + lane * 16;
    FragU wb0, wb1, wb2, wb3;
    load_pk(wb0, fb);
    load_pk(wb1, fb + 512);
    load_pk(wb2, fb + 1024);
    load_pk(wb3, fb + 1536);
    acc[0][0] = wmma_bf16(af0, wb0, acc[0][0]);
    acc[0][1] = wmma_bf16(af0, wb1, acc[0][1]);
    acc[0][2] = wmma_bf16(af0, wb2, acc[0][2]);
    acc[0][3] = wmma_bf16(af0, wb3, acc[0][3]);
    acc[1][0] = wmma_bf16(af1, wb0, acc[1][0]);
    acc[1][1] = wmma_bf16(af1, wb1, acc[1][1]);
    acc[1][2] = wmma_bf16(af1, wb2, acc[1][2]);
    acc[1][3] = wmma_bf16(af1, wb3, acc[1][3]);
  }

  // stage the 32x64 f32 tile to LDS (C layout: row = g*16 + hv*8+rr, col = t*16+r)
  float* tb = stile[w];
#pragma unroll
  for (int g = 0; g < 2; ++g)
#pragma unroll
    for (int rr = 0; rr < 8; ++rr) {
      float* tq = tb + (g * 16 + hv * 8 + rr) * 64 + r;
      tq[0]  = acc[g][0][rr];
      tq[16] = acc[g][1][rr];
      tq[32] = acc[g][2][rr];
      tq[48] = acc[g][3][rr];
    }
  __syncthreads();

  // RoPE + scatter: each lane owns one row (32 rotary pairs)
  const int m = mbase + lane;
  if (m < MTOT) {
    const int bi = m / N_SEQ, n = m % N_SEQ;
    const int tensor = sec / HEADS, h = sec % HEADS;
    const float* tr = stile[w] + lane * 64;
    if (tensor == 2) {  // V: transpose into feature-major, no RoPE
      unsigned short* base = Vt + (size_t)(bi * HEADS + h) * DIM_HEAD * NPAD + n;
#pragma unroll
      for (int c = 0; c < 64; ++c) base[(size_t)c * NPAD] = f2bf(tr[c]);
    } else {
      unsigned short* dst = (tensor == 0 ? Qp : Kp) +
                            ((size_t)(bi * HEADS + h) * NPAD + n) * DIM_HEAD;
      if (n == 0) {  // first token keeps original feature order
#pragma unroll
        for (int c = 0; c < 64; ++c) dst[c] = f2bf(tr[c]);
      } else {
        const size_t cidx = ((size_t)bi * (N_SEQ - 1) + (n - 1)) * 2;
        const float csum = coords[cidx] + coords[cidx + 1];
#pragma unroll
        for (int dd = 0; dd < 32; ++dd) {
          float ang = csum * exp2f(-13.0f * (float)dd / 32.0f);  // 8192 = 2^13
          float cs = __cosf(ang), sn = __sinf(ang);
          float i0 = tr[2 * dd], i1 = tr[2 * dd + 1];
          dst[dd]      = f2bf(i0 * cs - i1 * sn);
          dst[dd + 32] = f2bf(i0 * sn + i1 * cs);
        }
      }
    }
  }
}

// ---------------------------------------------------------------------------
// Flash attention: one wave per (16-query tile, head, batch).
// Per 32-key step: 4 WMMAs for S = Q K^T, f32 online softmax with 16-lane
// shfl_xor row reductions, P staged through LDS to build its A-fragment,
// then 4 WMMAs for the PV accumulation over the 64 output features.
// Alibi loads are index-clamped and selected with ternaries (no exec churn).
// ---------------------------------------------------------------------------
__global__ void __launch_bounds__(32)
attn_kernel(const unsigned short* __restrict__ Qp, const unsigned short* __restrict__ Kp,
            const unsigned short* __restrict__ Vt, const float* __restrict__ alibi,
            const unsigned char* __restrict__ mask, unsigned short* __restrict__ ao) {
  __shared__ alignas(16) unsigned short pbuf[16 * 32];
  const int lane = threadIdx.x & 31;
  const int hv = lane >> 4, r = lane & 15;
  const int qt = blockIdx.x, h = blockIdx.y, b = blockIdx.z;
  const int qbase = qt * 16;

  const unsigned short* qrow =
      Qp + ((size_t)(b * HEADS + h) * NPAD + qbase + r) * DIM_HEAD;
  FragU qa0, qa1;
  load_a2(qa0, qrow + 0 + hv * 16);
  load_a2(qa1, qrow + 32 + hv * 16);

  const unsigned short* kb = Kp + (size_t)(b * HEADS + h) * NPAD * DIM_HEAD;
  const unsigned short* vb = Vt + (size_t)(b * HEADS + h) * DIM_HEAD * NPAD;
  const float* ab_head = alibi + (size_t)h * N_SEQ * N_SEQ;

  float mrow[8], lrow[8], sc[8];
  v8f o0 = zero8(), o1 = zero8(), o2 = zero8(), o3 = zero8();
#pragma unroll
  for (int i = 0; i < 8; ++i) { mrow[i] = -__builtin_inff(); lrow[i] = 0.f; }

  for (int jt = 0; jt < NPAD / 32; ++jt) {
    const int jb = jt * 32;
    if (jb + 32 + lane < N_SEQ)  // pull next alibi tile toward L2/L0
      __builtin_prefetch(ab_head + (size_t)qbase * N_SEQ + jb + 32 + lane, 0, 1);

    // ---- S = Q K^T (keys jb..jb+31), f32 ----
    v8f c0 = zero8(), c1 = zero8();
    {
      const unsigned short* k0 = kb + (size_t)(jb + r) * DIM_HEAD + hv * 16;
      const unsigned short* k1 = kb + (size_t)(jb + 16 + r) * DIM_HEAD + hv * 16;
      FragU bk;
      load_pk(bk, k0);      c0 = wmma_bf16(qa0, bk, c0);
      load_pk(bk, k0 + 32); c0 = wmma_bf16(qa1, bk, c0);
      load_pk(bk, k1);      c1 = wmma_bf16(qa0, bk, c1);
      load_pk(bk, k1 + 32); c1 = wmma_bf16(qa1, bk, c1);
    }

    // ---- online softmax (branchless score path) ----
    const int j0 = jb + r, j1 = jb + 16 + r;
    const bool mj0 = (j0 < N_SEQ) && (mask[b * N_SEQ + j0] != 0);
    const bool mj1 = (j1 < N_SEQ) && (mask[b * N_SEQ + j1] != 0);
    const int jc0 = imin(j0, N_SEQ - 1), jc1 = imin(j1, N_SEQ - 1);
#pragma unroll
    for (int rr = 0; rr < 8; ++rr) {
      const int i = qbase + hv * 8 + rr;
      const float* ab = ab_head + (size_t)imin(i, N_SEQ - 1) * N_SEQ;
      const float bias0 = ab[jc0], bias1 = ab[jc1];
      const bool iok = (i < N_SEQ);
      float s0 = (iok && mj0) ? fmaf(c0[rr], ATT_SCALE, bias0) : -__builtin_inff();
      float s1 = (iok && mj1) ? fmaf(c1[rr], ATT_SCALE, bias1) : -__builtin_inff();
      float mx = fmaxf(s0, s1);
#pragma unroll
      for (int o = 8; o > 0; o >>= 1) mx = fmaxf(mx, __shfl_xor(mx, o, 32));
      const float mnew = fmaxf(mrow[rr], mx);
      const bool dead = (mnew == -__builtin_inff());
      const float scl = dead ? 1.f : __expf(mrow[rr] - mnew);
      const float p0 = dead ? 0.f : __expf(s0 - mnew);
      const float p1 = dead ? 0.f : __expf(s1 - mnew);
      float rsum = p0 + p1;
#pragma unroll
      for (int o = 8; o > 0; o >>= 1) rsum += __shfl_xor(rsum, o, 32);
      lrow[rr] = lrow[rr] * scl + rsum;
      mrow[rr] = mnew;
      sc[rr] = scl;
      pbuf[(hv * 8 + rr) * 32 + r]      = f2bf(p0);
      pbuf[(hv * 8 + rr) * 32 + 16 + r] = f2bf(p1);
    }
#pragma unroll
    for (int rr = 0; rr < 8; ++rr) {
      o0[rr] *= sc[rr]; o1[rr] *= sc[rr]; o2[rr] *= sc[rr]; o3[rr] *= sc[rr];
    }
    __syncthreads();

    // ---- O += P V ----
    FragU pa;
    load_a2(pa, pbuf + r * 32 + hv * 16);
    FragU bv;
    load_pk(bv, vb + (size_t)(0 * 16 + r) * NPAD + jb + hv * 16); o0 = wmma_bf16(pa, bv, o0);
    load_pk(bv, vb + (size_t)(1 * 16 + r) * NPAD + jb + hv * 16); o1 = wmma_bf16(pa, bv, o1);
    load_pk(bv, vb + (size_t)(2 * 16 + r) * NPAD + jb + hv * 16); o2 = wmma_bf16(pa, bv, o2);
    load_pk(bv, vb + (size_t)(3 * 16 + r) * NPAD + jb + hv * 16); o3 = wmma_bf16(pa, bv, o3);
    __syncthreads();
  }

#pragma unroll
  for (int rr = 0; rr < 8; ++rr) {
    const int i = qbase + hv * 8 + rr;
    if (i < N_SEQ) {
      const float inv = 1.f / lrow[rr];
      unsigned short* dst = ao + ((size_t)b * N_SEQ + i) * DIM + h * DIM_HEAD + r;
      dst[0]  = f2bf(o0[rr] * inv);
      dst[16] = f2bf(o1[rr] * inv);
      dst[32] = f2bf(o2[rr] * inv);
      dst[48] = f2bf(o3[rr] * inv);
    }
  }
}

// ---------------------------------------------------------------------------
// Output projection GEMM (8200x768)x(768x768), f32 result straight to d_out.
// Grid (257, 3), block 128 (4 waves); wave = one 32x64 tile, clamped A rows.
// ---------------------------------------------------------------------------
__global__ void __launch_bounds__(128)
outproj_kernel(const unsigned short* __restrict__ ao, const unsigned short* __restrict__ wo,
               float* __restrict__ out) {
  const int w = threadIdx.x >> 5, lane = threadIdx.x & 31;
  const int hv = lane >> 4, r = lane & 15;
  const int sec = blockIdx.y * 4 + w;  // 0..11
  const int cbase = sec * 64;
  const int mbase = blockIdx.x * 32;

  v8f acc[2][4];
#pragma unroll
  for (int g = 0; g < 2; ++g)
#pragma unroll
    for (int t = 0; t < 4; ++t) acc[g][t] = zero8();

  const unsigned short* arow0 = ao + (size_t)imin(mbase + r,      MTOT - 1) * DIM;
  const unsigned short* arow1 = ao + (size_t)imin(mbase + 16 + r, MTOT - 1) * DIM;

  for (int kt = 0; kt < DIM / 32; ++kt) {
    FragU af0, af1;
    load_a2(af0, arow0 + kt * 32 + hv * 16);
    load_a2(af1, arow1 + kt * 32 + hv * 16);
    const unsigned short* fb = wo + ((size_t)kt * 48 + (cbase >> 4)) * 512 + lane * 16;
    FragU wb0, wb1, wb2, wb3;
    load_pk(wb0, fb);
    load_pk(wb1, fb + 512);
    load_pk(wb2, fb + 1024);
    load_pk(wb3, fb + 1536);
    acc[0][0] = wmma_bf16(af0, wb0, acc[0][0]);
    acc[0][1] = wmma_bf16(af0, wb1, acc[0][1]);
    acc[0][2] = wmma_bf16(af0, wb2, acc[0][2]);
    acc[0][3] = wmma_bf16(af0, wb3, acc[0][3]);
    acc[1][0] = wmma_bf16(af1, wb0, acc[1][0]);
    acc[1][1] = wmma_bf16(af1, wb1, acc[1][1]);
    acc[1][2] = wmma_bf16(af1, wb2, acc[1][2]);
    acc[1][3] = wmma_bf16(af1, wb3, acc[1][3]);
  }
#pragma unroll
  for (int g = 0; g < 2; ++g)
#pragma unroll
    for (int rr = 0; rr < 8; ++rr) {
      const int m = mbase + g * 16 + hv * 8 + rr;
      if (m < MTOT) {
        float* orow = out + (size_t)m * DIM + cbase + r;
        orow[0]  = acc[g][0][rr];
        orow[16] = acc[g][1][rr];
        orow[32] = acc[g][2][rr];
        orow[48] = acc[g][3][rr];
      }
    }
}

// ---------------------------------------------------------------------------
extern "C" void kernel_launch(void* const* d_in, const int* in_sizes, int n_in,
                              void* d_out, int out_size, void* d_ws, size_t ws_size,
                              hipStream_t stream) {
  (void)in_sizes; (void)n_in; (void)out_size; (void)ws_size;
  const float* x            = (const float*)d_in[0];
  const float* alibi        = (const float*)d_in[1];
  const float* coords       = (const float*)d_in[2];
  const unsigned char* mask = (const unsigned char*)d_in[3];
  const float* gamma        = (const float*)d_in[4];
  const float* beta         = (const float*)d_in[5];
  const float* Wqkv         = (const float*)d_in[6];
  const float* Wout         = (const float*)d_in[7];
  float* out = (float*)d_out;

  char* p = (char*)d_ws;
  auto alloc = [&](size_t bytes) -> void* {
    void* r = (void*)p;
    p += (bytes + 255) & ~(size_t)255;
    return r;
  };
  unsigned short* xn  = (unsigned short*)alloc((size_t)MTOT * DIM * 2);
  unsigned short* wqp = (unsigned short*)alloc((size_t)DIM * 3 * DIM * 2);
  unsigned short* wop = (unsigned short*)alloc((size_t)DIM * DIM * 2);
  unsigned short* Qp  = (unsigned short*)alloc((size_t)B_SZ * HEADS * NPAD * DIM_HEAD * 2);
  unsigned short* Kp  = (unsigned short*)alloc((size_t)B_SZ * HEADS * NPAD * DIM_HEAD * 2);
  unsigned short* Vt  = (unsigned short*)alloc((size_t)B_SZ * HEADS * NPAD * DIM_HEAD * 2);
  unsigned short* ao  = (unsigned short*)alloc((size_t)MTOT * DIM * 2);

  pack_w_kernel<<<(DIM * 3 * DIM + 255) / 256, 256, 0, stream>>>(Wqkv, wqp, 3 * DIM);
  pack_w_kernel<<<(DIM * DIM + 255) / 256, 256, 0, stream>>>(Wout, wop, DIM);
  ln_kernel<<<MTOT, 256, 0, stream>>>(x, gamma, beta, xn);
  qkv_kernel<<<dim3(MTILES32, 9), 128, 0, stream>>>(xn, wqp, coords, Qp, Kp, Vt);
  attn_kernel<<<dim3(65, HEADS, B_SZ), 32, 0, stream>>>(Qp, Kp, Vt, alibi, mask, ao);
  outproj_kernel<<<dim3(MTILES32, 3), 128, 0, stream>>>(ao, wop, out);
}
